// InterWindowAttn_40381282517305
// MI455X (gfx1250) — matrix operand order repaired and checked by reading, not compiled, not run on hardware
//
#include <hip/hip_runtime.h>

// ---------------------------------------------------------------------------
// Problem constants (fixed by setup_inputs)
// ---------------------------------------------------------------------------
#define GSW   8                 // window side
#define PIX   64                // GSW*GSW pixels per window
#define CC    128               // channels
#define BB    4
#define HH    256
#define WWI   256
#define GHN   32                // HH/GSW
#define GWN   32
#define NWIN  4096              // BB*GHN*GWN
#define TOPK  3
#define PITCH 136               // padded f16 pitch (dodge LDS bank conflicts)
#define WPB   8                 // windows per persistent block
#define NBLK  (NWIN / WPB)      // 512 persistent blocks

typedef __attribute__((ext_vector_type(16))) _Float16     v16h;
typedef __attribute__((ext_vector_type(8)))  float        v8f;
typedef __attribute__((ext_vector_type(4)))  unsigned int v4u;
typedef __attribute__((ext_vector_type(8)))  unsigned int v8u;

// ---------------------------------------------------------------------------
// WMMA helpers: v_wmma_f32_16x16x32_f16, D = A(16x32 f16) * B(32x16 f16) + C
// ---------------------------------------------------------------------------
static __device__ inline v8f wmma_step(v16h a, v16h b, v8f c) {
  return __builtin_amdgcn_wmma_f32_16x16x32_f16(
      false, a, false, b, (short)0, c, false, false);
}

// A fragment: 16x32 (MxK), row-major f16 matrix in LDS with pitch `pa`.
static __device__ inline v16h load_a_frag(const _Float16* sm, int pa, int m0, int k0) {
  const int lane  = threadIdx.x & 31;
  const int khalf = (lane >> 4) << 4;
  const _Float16* row = sm + (m0 + (lane & 15)) * pa;
  v16h a;
#pragma unroll
  for (int e = 0; e < 16; ++e) {
    int k = k0 + e + khalf;
    a[e] = row[k];
  }
  return a;
}

// B fragment from an N-major matrix (rows = N, cols = K): B[k][n] = sm[n*pn + k].
static __device__ inline v16h load_b_frag_nmajor(const _Float16* sm, int pn, int k0, int n0) {
  const int lane = threadIdx.x & 31;
  const _Float16* row = sm + (n0 + (lane & 15)) * pn;
  const int kb = k0 + ((lane >> 4) << 4);
  v16h b;
#pragma unroll
  for (int e = 0; e < 16; ++e) b[e] = row[kb + e];
  return b;
}

// B fragment from a K-major matrix (rows = K, cols = N): B[k][n] = sm[k*pn + n].
static __device__ inline v16h load_b_frag_kmajor(const _Float16* sm, int pn, int k0, int n0) {
  const int lane = threadIdx.x & 31;
  const int n = n0 + (lane & 15);
  const int kb = k0 + ((lane >> 4) << 4);
  v16h b;
#pragma unroll
  for (int e = 0; e < 16; ++e) b[e] = sm[(kb + e) * pn + n];
  return b;
}

// ---------------------------------------------------------------------------
// Tensor Data Mover: stage a dense 128x128 f16 matrix from global into LDS,
// padding 4 DWORDs after every 64 DWORDs (row of 128 f16) -> pitch 136 halves.
// D# layout per CDNA5 ISA 08_async_tensor §8.3/8.4. 2D tile -> 2 SGPR groups.
// ---------------------------------------------------------------------------
static __device__ inline void tdm_load_weight_128x128(const _Float16* gsrc,
                                                      unsigned lds_byte_off) {
  const unsigned long long ga = (unsigned long long)(const void*)gsrc;
  v4u g0;
  g0[0] = 1u;                                            // count=1, user mode
  g0[1] = lds_byte_off;                                  // lds_addr
  g0[2] = (unsigned)(ga & 0xffffffffu);                  // global_addr[31:0]
  g0[3] = (unsigned)((ga >> 32) & 0x1ffffffu)            // global_addr[56:32]
          | (2u << 30);                                  // type=2 ("image")
  v8u g1;
  g1[0] = (1u << 16)        // data_size = 1 -> 2 bytes
        | (1u << 20)        // pad_enable
        | (5u << 22)        // pad_interval: code 5 -> every 64 DWORDs
        | (3u << 25);       // pad_amount: code 3 -> 4 DWORDs (8 halves)
  g1[1] = (128u << 16);     // tensor_dim0 = 128 (bits 79:48, low part)
  g1[2] = (128u << 16);     // tensor_dim1 = 128 (bits 111:80, low part)
  g1[3] = (128u << 16);     // tile_dim0 = 128 (bits 127:112)
  g1[4] = 128u;             // tile_dim1 = 128 (bits 143:128)
  g1[5] = 128u;             // tensor_dim0_stride = 128 (bits 207:160, low)
  g1[6] = 0u;
  g1[7] = 0u;
  asm volatile("tensor_load_to_lds %0, %1" :: "s"(g0), "s"(g1) : "memory");
}

// ---------------------------------------------------------------------------
// Kernel 0: convert the four 128x128 f32 weight matrices to dense f16 in ws.
// Order: Wq, Wk, Wv, Wp at wh + k*16384.
// ---------------------------------------------------------------------------
__global__ __launch_bounds__(256) void k_cvtw(const float* __restrict__ wq,
                                              const float* __restrict__ wk,
                                              const float* __restrict__ wv,
                                              const float* __restrict__ wp,
                                              _Float16* __restrict__ wh) {
  const int i = blockIdx.x * 256 + threadIdx.x;   // 0 .. 4*16384-1
  const int m = i >> 14, r = i & 16383;
  const float* src = (m == 0) ? wq : (m == 1) ? wk : (m == 2) ? wv : wp;
  wh[i] = (_Float16)src[r];
}

// ---------------------------------------------------------------------------
// Kernel 1: adaptive max pool 8x8 -> dsx f16 [NWIN][CC]
// ---------------------------------------------------------------------------
__global__ __launch_bounds__(256) void k_pool(const float* __restrict__ x,
                                              _Float16* __restrict__ dsx) {
  __shared__ float sCol[256];
  const int blk = blockIdx.x;
  const int b   = blk >> 12;
  const int rem = blk & 4095;
  const int c   = rem >> 5;
  const int gy  = rem & 31;
  const int t   = threadIdx.x;
  const float* base = x + (((size_t)b * CC + c) * HH + gy * GSW) * WWI + t;
  float m = -3.4e38f;
#pragma unroll
  for (int r = 0; r < GSW; ++r) m = fmaxf(m, base[r * WWI]);
  sCol[t] = m;
  __syncthreads();
  if (t < GWN) {
    float mm = -3.4e38f;
#pragma unroll
    for (int u = 0; u < GSW; ++u) mm = fmaxf(mm, sCol[t * GSW + u]);
    const int n = b * (GHN * GWN) + gy * GWN + t;
    dsx[n * CC + c] = (_Float16)mm;
  }
}

// ---------------------------------------------------------------------------
// Kernel 2: sim = dsx * dsx^T  [NWIN x NWIN] f32, WMMA 64x64 tiles.
// ---------------------------------------------------------------------------
__global__ __launch_bounds__(128) void k_sim(const _Float16* __restrict__ dsx,
                                             float* __restrict__ sim) {
  __shared__ _Float16 sA[64 * PITCH];
  __shared__ _Float16 sB[64 * PITCH];
  const int bi = blockIdx.x >> 6;
  const int bj = blockIdx.x & 63;
  const int rowbase = bi * 64, colbase = bj * 64;
  for (int i = threadIdx.x; i < 64 * CC; i += 128) {
    const int r = i >> 7, c = i & 127;
    sA[r * PITCH + c] = dsx[(rowbase + r) * CC + c];
    sB[r * PITCH + c] = dsx[(colbase + r) * CC + c];
  }
  __syncthreads();
  const int wave = threadIdx.x >> 5;
  const int lane = threadIdx.x & 31;
  const int m0 = wave * 16;
  for (int nt = 0; nt < 4; ++nt) {
    const int n0 = nt * 16;
    v8f acc;
#pragma unroll
    for (int r = 0; r < 8; ++r) acc[r] = 0.0f;
#pragma unroll
    for (int k0 = 0; k0 < CC; k0 += 32) {
      v16h a = load_a_frag(sA, PITCH, m0, k0);
      v16h b = load_b_frag_nmajor(sB, PITCH, k0, n0);
      acc = wmma_step(a, b, acc);
    }
#pragma unroll
    for (int r = 0; r < 8; ++r) {
      const int m = m0 + r + ((lane >> 4) << 3);
      const int nn = n0 + (lane & 15);
      sim[(size_t)(rowbase + m) * NWIN + (colbase + nn)] = acc[r];
    }
  }
}

// ---------------------------------------------------------------------------
// Kernel 3: global top-3 per window + softmax weights (sim symmetric ->
// coalesced column scan).
// ---------------------------------------------------------------------------
__global__ __launch_bounds__(256) void k_topk(const float* __restrict__ sim,
                                              float* __restrict__ topw,
                                              int* __restrict__ topi) {
  const int col = blockIdx.x * 256 + threadIdx.x;
  float v0 = -3.4e38f, v1 = -3.4e38f, v2 = -3.4e38f;
  int i0 = 0, i1 = 0, i2 = 0;
  for (int j = 0; j < NWIN; ++j) {
    const float v = sim[(size_t)j * NWIN + col];
    if (v > v0)      { v2 = v1; i2 = i1; v1 = v0; i1 = i0; v0 = v; i0 = j; }
    else if (v > v1) { v2 = v1; i2 = i1; v1 = v;  i1 = j; }
    else if (v > v2) { v2 = v;  i2 = j; }
  }
  const float e1 = __expf(v1 - v0), e2 = __expf(v2 - v0);
  const float inv = 1.0f / (1.0f + e1 + e2);
  topw[col * TOPK + 0] = inv;
  topw[col * TOPK + 1] = e1 * inv;
  topw[col * TOPK + 2] = e2 * inv;
  topi[col * TOPK + 0] = i0;
  topi[col * TOPK + 1] = i1;
  topi[col * TOPK + 2] = i2;
}

// ---------------------------------------------------------------------------
// Kernel 4: persistent fused attention. 512 blocks x 8 windows, 256 thr (8 waves).
// Weights staged ONCE per block via TDM (pitch-padded by TDM's pad feature).
// LDS budget: 5*17408 + 16384 + 4*34816 = 242,688 B of the 320 KB WGP pool.
// ---------------------------------------------------------------------------
#define OFF_XW  0
#define OFF_CTX (OFF_XW  + PIX * PITCH * 2)
#define OFF_Q   (OFF_CTX + PIX * PITCH * 2)
#define OFF_K   (OFF_Q   + PIX * PITCH * 2)
#define OFF_V   (OFF_K   + PIX * PITCH * 2)
#define OFF_LOG (OFF_V   + PIX * PITCH * 2)
#define OFF_WQ  (OFF_LOG + PIX * PIX * 4)
#define OFF_WK  (OFF_WQ  + CC * PITCH * 2)
#define OFF_WV  (OFF_WK  + CC * PITCH * 2)
#define OFF_WP  (OFF_WV  + CC * PITCH * 2)
#define SMEM_BYTES (OFF_WP + CC * PITCH * 2)   // 242688 B

// 64x128 = A[64xK=128] * B(nmajor: B[k=c][n=o] = sB[o][c]) + bias -> f16 LDS
static __device__ inline void gemm128_bias_to_h(const _Float16* sA, const _Float16* sB,
                                                const float* __restrict__ bias,
                                                _Float16* sD) {
  const int lane = threadIdx.x & 31;
  const int wave = threadIdx.x >> 5;
  const int n0 = wave * 16;
  const float bv = bias[n0 + (lane & 15)];
#pragma unroll
  for (int mt = 0; mt < 4; ++mt) {
    const int m0 = mt * 16;
    v8f acc;
#pragma unroll
    for (int r = 0; r < 8; ++r) acc[r] = bv;
#pragma unroll
    for (int k0 = 0; k0 < CC; k0 += 32) {
      v16h a = load_a_frag(sA, PITCH, m0, k0);
      v16h b = load_b_frag_nmajor(sB, PITCH, k0, n0);
      acc = wmma_step(a, b, acc);
    }
#pragma unroll
    for (int r = 0; r < 8; ++r) {
      const int m = m0 + r + ((lane >> 4) << 3);
      sD[m * PITCH + n0 + (lane & 15)] = (_Float16)acc[r];
    }
  }
}

__global__ __launch_bounds__(256) void k_attn(
    const float* __restrict__ x,
    const _Float16* __restrict__ wh,          // f16 Wq|Wk|Wv|Wp, 16384 each
    const float* __restrict__ bq, const float* __restrict__ bk,
    const float* __restrict__ bv, const float* __restrict__ bp,
    const float* __restrict__ lepe_w, const float* __restrict__ lepe_b,
    const float* __restrict__ topw, const int* __restrict__ topi,
    float* __restrict__ out) {
  extern __shared__ char smem[];
  _Float16* sXW  = (_Float16*)(smem + OFF_XW);
  _Float16* sCTX = (_Float16*)(smem + OFF_CTX);
  _Float16* sQ   = (_Float16*)(smem + OFF_Q);
  _Float16* sK   = (_Float16*)(smem + OFF_K);
  _Float16* sV   = (_Float16*)(smem + OFF_V);
  float*    sLOG = (float*)(smem + OFF_LOG);
  _Float16* sWQ  = (_Float16*)(smem + OFF_WQ);
  _Float16* sWK  = (_Float16*)(smem + OFF_WK);
  _Float16* sWV  = (_Float16*)(smem + OFF_WV);
  _Float16* sWP  = (_Float16*)(smem + OFF_WP);

  const int tid  = threadIdx.x;
  const int lane = tid & 31;
  const int wave = tid >> 5;

  // ---- stage all four weight matrices once per block via TDM ---------------
  if (wave == 0) {
    tdm_load_weight_128x128(wh + 0 * 16384, OFF_WQ);
    tdm_load_weight_128x128(wh + 1 * 16384, OFF_WK);
    tdm_load_weight_128x128(wh + 2 * 16384, OFF_WV);
    tdm_load_weight_128x128(wh + 3 * 16384, OFF_WP);
    __builtin_amdgcn_s_wait_tensorcnt(0);
  }
  __syncthreads();

  for (int w = 0; w < WPB; ++w) {
    const int n  = blockIdx.x * WPB + w;
    const int b  = n >> 10;
    const int gy = (n >> 5) & 31;
    const int gx = n & 31;

    // ---- window pixels (f16) + fused neighbor context ----------------------
    float w3[TOPK];
    int   m3[TOPK];
#pragma unroll
    for (int k = 0; k < TOPK; ++k) {
      w3[k] = topw[n * TOPK + k];
      m3[k] = topi[n * TOPK + k];
    }
    for (int i = tid; i < PIX * CC; i += 256) {
      const int c = i >> 6, p = i & 63;
      const int py = p >> 3, px = p & 7;
      const float xv =
          x[(((size_t)b * CC + c) * HH + gy * GSW + py) * WWI + gx * GSW + px];
      sXW[p * PITCH + c] = (_Float16)xv;
      float acc = 0.0f;
#pragma unroll
      for (int k = 0; k < TOPK; ++k) {
        const int m  = m3[k];
        const int nb = m >> 10, ngy = (m >> 5) & 31, ngx = m & 31;
        acc += w3[k] *
               x[(((size_t)nb * CC + c) * HH + ngy * GSW + py) * WWI + ngx * GSW + px];
      }
      sCTX[p * PITCH + c] = (_Float16)acc;
    }
    __syncthreads();

    // ---- Q/K/V GEMMs back-to-back (disjoint outputs, one barrier) ----------
    gemm128_bias_to_h(sXW,  sWQ, bq, sQ);
    gemm128_bias_to_h(sCTX, sWK, bk, sK);
    gemm128_bias_to_h(sCTX, sWV, bv, sV);
    __syncthreads();

    // ---- logits = Q @ K^T * scale  (64x64, K=128) --------------------------
    {
      const int n0 = (wave & 3) * 16;
#pragma unroll
      for (int half = 0; half < 2; ++half) {
        const int m0 = ((wave >> 2) * 2 + half) * 16;
        v8f acc;
#pragma unroll
        for (int r = 0; r < 8; ++r) acc[r] = 0.0f;
#pragma unroll
        for (int k0 = 0; k0 < CC; k0 += 32) {
          v16h a   = load_a_frag(sQ, PITCH, m0, k0);
          v16h bfr = load_b_frag_nmajor(sK, PITCH, k0, n0);
          acc = wmma_step(a, bfr, acc);
        }
        const float scale = 0.088388347648318447f;   // 128^-0.5
#pragma unroll
        for (int r = 0; r < 8; ++r) {
          const int m = m0 + r + ((lane >> 4) << 3);
          sLOG[m * PIX + n0 + (lane & 15)] = acc[r] * scale;
        }
      }
    }
    __syncthreads();

    // ---- softmax rows -> attn (f16) into sCTX (reused) ---------------------
    if (tid < PIX) {
      float mx = -3.4e38f;
#pragma unroll 4
      for (int j = 0; j < PIX; ++j) mx = fmaxf(mx, sLOG[tid * PIX + j]);
      float s = 0.0f;
      float e[PIX];
#pragma unroll 4
      for (int j = 0; j < PIX; ++j) { e[j] = __expf(sLOG[tid * PIX + j] - mx); s += e[j]; }
      const float inv = 1.0f / s;
#pragma unroll 4
      for (int j = 0; j < PIX; ++j) sCTX[tid * PITCH + j] = (_Float16)(e[j] * inv);
    }
    __syncthreads();

    // ---- attn @ V + LePE -> f16 into sQ (reused as OUT) --------------------
    {
      const int n0 = wave * 16;
      const int cch = n0 + (lane & 15);
      float lw[9];
#pragma unroll
      for (int j = 0; j < 9; ++j) lw[j] = lepe_w[cch * 9 + j];
      const float lb = lepe_b[cch];
#pragma unroll
      for (int mt = 0; mt < 4; ++mt) {
        const int m0 = mt * 16;
        v8f acc;
#pragma unroll
        for (int r = 0; r < 8; ++r) acc[r] = 0.0f;
#pragma unroll
        for (int k0 = 0; k0 < PIX; k0 += 32) {
          v16h a   = load_a_frag(sCTX, PITCH, m0, k0);
          v16h bfr = load_b_frag_kmajor(sV, PITCH, k0, n0);
          acc = wmma_step(a, bfr, acc);
        }
#pragma unroll
        for (int r = 0; r < 8; ++r) {
          const int m = m0 + r + ((lane >> 4) << 3);
          const int py = m >> 3, px = m & 7;
          float l = lb;
#pragma unroll
          for (int dy = 0; dy < 3; ++dy) {
#pragma unroll
            for (int dx = 0; dx < 3; ++dx) {
              const int yy = py + dy - 1, xx = px + dx - 1;
              if (yy >= 0 && yy < GSW && xx >= 0 && xx < GSW)
                l += lw[dy * 3 + dx] * (float)sXW[(yy * GSW + xx) * PITCH + cch];
            }
          }
          sQ[m * PITCH + cch] = (_Float16)(acc[r] + l);
        }
      }
    }
    __syncthreads();

    // ---- out = OUT @ Wp^T + bp, scattered store = image reassembly ---------
    {
      const int n0 = wave * 16;
      const int cch = n0 + (lane & 15);
      const float bvp = bp[cch];
#pragma unroll
      for (int mt = 0; mt < 4; ++mt) {
        const int m0 = mt * 16;
        v8f acc;
#pragma unroll
        for (int r = 0; r < 8; ++r) acc[r] = bvp;
#pragma unroll
        for (int k0 = 0; k0 < CC; k0 += 32) {
          v16h a   = load_a_frag(sQ, PITCH, m0, k0);
          v16h bfr = load_b_frag_nmajor(sWP, PITCH, k0, n0);
          acc = wmma_step(a, bfr, acc);
        }
#pragma unroll
        for (int r = 0; r < 8; ++r) {
          const int m = m0 + r + ((lane >> 4) << 3);
          const int py = m >> 3, px = m & 7;
          out[(((size_t)b * CC + cch) * HH + gy * GSW + py) * WWI + gx * GSW + px] = acc[r];
        }
      }
    }
    // No trailing barrier needed: next iteration's LDS writes are ordered by
    // its own post-load barrier, and all readers of this iteration's buffers
    // passed the pre-proj barrier.
  }
}

// ---------------------------------------------------------------------------
// Host launcher. Workspace layout (bytes):
//   [0, 1MB)     dsx f16 [NWIN][CC]
//   [+64MB)      sim f32 [NWIN][NWIN]
//   topw f32 [NWIN][3], topi i32 [NWIN][3], wh f16 [4][128][128]
// ---------------------------------------------------------------------------
extern "C" void kernel_launch(void* const* d_in, const int* in_sizes, int n_in,
                              void* d_out, int out_size, void* d_ws, size_t ws_size,
                              hipStream_t stream) {
  (void)in_sizes; (void)n_in; (void)out_size; (void)ws_size;
  const float* x      = (const float*)d_in[0];
  const float* wq     = (const float*)d_in[1];
  const float* bq     = (const float*)d_in[2];
  const float* wk     = (const float*)d_in[3];
  const float* bk     = (const float*)d_in[4];
  const float* wv     = (const float*)d_in[5];
  const float* bvv    = (const float*)d_in[6];
  const float* wp     = (const float*)d_in[7];
  const float* bp     = (const float*)d_in[8];
  const float* lepe_w = (const float*)d_in[9];
  const float* lepe_b = (const float*)d_in[10];
  float* out = (float*)d_out;

  char* ws = (char*)d_ws;
  const size_t OFF_DSX  = 0;
  const size_t OFF_SIM  = OFF_DSX + (size_t)NWIN * CC * sizeof(_Float16);
  const size_t OFF_TOPW = OFF_SIM + (size_t)NWIN * NWIN * sizeof(float);
  const size_t OFF_TOPI = OFF_TOPW + (size_t)NWIN * TOPK * sizeof(float);
  const size_t OFF_WH   = OFF_TOPI + (size_t)NWIN * TOPK * sizeof(int);
  _Float16* dsx  = (_Float16*)(ws + OFF_DSX);
  float*    sim  = (float*)(ws + OFF_SIM);
  float*    topw = (float*)(ws + OFF_TOPW);
  int*      topi = (int*)(ws + OFF_TOPI);
  _Float16* wh   = (_Float16*)(ws + OFF_WH);

  k_cvtw<<<dim3(4 * CC * CC / 256), dim3(256), 0, stream>>>(wq, wk, wv, wp, wh);
  k_pool<<<dim3(BB * CC * GHN), dim3(256), 0, stream>>>(x, dsx);
  k_sim<<<dim3((NWIN / 64) * (NWIN / 64)), dim3(128), 0, stream>>>(dsx, sim);
  k_topk<<<dim3(NWIN / 256), dim3(256), 0, stream>>>(sim, topw, topi);
  k_attn<<<dim3(NBLK), dim3(256), SMEM_BYTES, stream>>>(
      x, wh, bq, bk, bvv, bp, lepe_w, lepe_b, topw, topi, out);
}
